// StreamingSelfAttention_89962384982154
// MI455X (gfx1250) — matrix operand order repaired and checked by reading, not compiled, not run on hardware
//
#include <hip/hip_runtime.h>

typedef __bf16 bf16;
typedef __attribute__((ext_vector_type(16))) __bf16 v16bf;
typedef __attribute__((ext_vector_type(4)))  __bf16 v4bf;
typedef __attribute__((ext_vector_type(8)))  float  v8f;
typedef __attribute__((ext_vector_type(4)))  float  v4f;

#define WMMA_BF16(a, b, c) \
  __builtin_amdgcn_wmma_f32_16x16x32_bf16(false, (a), false, (b), (short)0, (c), false, false)

constexpr int Bx = 4, S = 2048, D = 1024, H = 16, HD = 64;
constexpr int M  = Bx * S;       // 8192 rows
constexpr int N3 = 3 * D;        // 3072
constexpr int LOOK = 64, LCTX = 512;

// Load a 16x32 A-fragment or 32x16 B-fragment as two 16-byte chunks.
__device__ __forceinline__ v16bf load2x16(const bf16* p0, const bf16* p1) {
  union { v16bf v; v4f f[2]; } u;
  u.f[0] = *reinterpret_cast<const v4f*>(p0);
  u.f[1] = *reinterpret_cast<const v4f*>(p1);
  return u.v;
}

// ---------------- fp32 -> bf16 conversion ----------------
__global__ void cvt_bf16(const float* __restrict__ in, bf16* __restrict__ out, int n4) {
  int i = blockIdx.x * blockDim.x + threadIdx.x;
  if (i < n4) {
    v4f x = reinterpret_cast<const v4f*>(in)[i];
    v4bf y;
    y[0] = (bf16)x[0]; y[1] = (bf16)x[1]; y[2] = (bf16)x[2]; y[3] = (bf16)x[3];
    reinterpret_cast<v4bf*>(out)[i] = y;
  }
}

// ---------------- QKV projection: [M,D] @ W_in^T + b_in -> Q,K,V^T ----------------
// 32x64 tile per wave; K loop unrolled x2 with ping-pong fragment sets (no copies).
__global__ __launch_bounds__(256) void gemm_qkv(
    const bf16* __restrict__ xb, const bf16* __restrict__ wi,
    const float* __restrict__ bin,
    bf16* __restrict__ qo, bf16* __restrict__ ko, bf16* __restrict__ vto) {
  const int lane = threadIdx.x & 31;
  const int wid  = (blockIdx.x * blockDim.x + threadIdx.x) >> 5;
  const int NT = N3 / 64;                 // 48 column tiles
  const int mt = wid / NT, nt = wid % NT;
  const int l15 = lane & 15, kh = lane >> 4;

  const bf16* arow0 = xb + (mt * 32 + l15) * D;
  const bf16* arow1 = arow0 + 16 * D;
  const bf16* br[4];
#pragma unroll
  for (int j = 0; j < 4; ++j) br[j] = wi + (nt * 64 + j * 16 + l15) * D;

  v8f acc[2][4] = {};
  v16bf a0A, a1A, bA[4], a0B, a1B, bB[4];
  // prologue: set A <- k = 0
  a0A = load2x16(arow0 + kh * 8, arow0 + 16 + kh * 8);
  a1A = load2x16(arow1 + kh * 8, arow1 + 16 + kh * 8);
#pragma unroll
  for (int j = 0; j < 4; ++j) bA[j] = load2x16(br[j] + kh * 8, br[j] + 16 + kh * 8);

  for (int k0 = 0; k0 < D; k0 += 64) {
    const int k1 = k0 + 32;                       // always < D
    a0B = load2x16(arow0 + k1 + kh * 8, arow0 + k1 + 16 + kh * 8);
    a1B = load2x16(arow1 + k1 + kh * 8, arow1 + k1 + 16 + kh * 8);
#pragma unroll
    for (int j = 0; j < 4; ++j)
      bB[j] = load2x16(br[j] + k1 + kh * 8, br[j] + k1 + 16 + kh * 8);
#pragma unroll
    for (int j = 0; j < 4; ++j) {
      acc[0][j] = WMMA_BF16(a0A, bA[j], acc[0][j]);
      acc[1][j] = WMMA_BF16(a1A, bA[j], acc[1][j]);
    }
    const int k2 = (k0 + 64 < D) ? (k0 + 64) : 0; // last iter: dummy valid reload
    a0A = load2x16(arow0 + k2 + kh * 8, arow0 + k2 + 16 + kh * 8);
    a1A = load2x16(arow1 + k2 + kh * 8, arow1 + k2 + 16 + kh * 8);
#pragma unroll
    for (int j = 0; j < 4; ++j)
      bA[j] = load2x16(br[j] + k2 + kh * 8, br[j] + k2 + 16 + kh * 8);
#pragma unroll
    for (int j = 0; j < 4; ++j) {
      acc[0][j] = WMMA_BF16(a0B, bB[j], acc[0][j]);
      acc[1][j] = WMMA_BF16(a1B, bB[j], acc[1][j]);
    }
  }
#pragma unroll
  for (int j = 0; j < 4; ++j) {
    int n = nt * 64 + j * 16 + l15;
    float bias = bin[n];
    int which = n >> 10, dd = n & (D - 1), h = dd >> 6, e = dd & 63;
#pragma unroll
    for (int i = 0; i < 2; ++i)
#pragma unroll
      for (int r = 0; r < 8; ++r) {
        int m = mt * 32 + i * 16 + kh * 8 + r;
        int bb = m >> 11, s = m & (S - 1);
        bf16 val = (bf16)(acc[i][j][r] + bias);
        if (which == 0)      qo[((bb * H + h) * S + s) * HD + e] = val;
        else if (which == 1) ko[((bb * H + h) * S + s) * HD + e] = val;
        else                 vto[((bb * H + h) * HD + e) * S + s] = val;  // V transposed
      }
  }
}

// ---------------- banded flash attention, one wave per 16-query tile ----------------
__global__ __launch_bounds__(256) void attn(
    const bf16* __restrict__ qb, const bf16* __restrict__ kbuf,
    const bf16* __restrict__ vtb, bf16* __restrict__ ctx) {
  __shared__ __align__(16) bf16 lds[8][16][32];   // per-wave P staging tile
  const int lane = threadIdx.x & 31;
  const int wl   = threadIdx.x >> 5;
  const int wid  = blockIdx.x * 8 + wl;
  const int s0 = (wid & (S / 16 - 1)) << 4;       // query tile start
  const int h  = (wid >> 7) & (H - 1);
  const int bb = wid >> 11;
  const int l15 = lane & 15, kh = lane >> 4;

  const bf16* qh = qb   + (size_t)((bb * H + h) * S) * HD;
  const bf16* kp = kbuf + (size_t)((bb * H + h) * S) * HD;
  const bf16* vh = vtb  + (size_t)((bb * H + h) * HD) * S;

  const bf16* qrow = qh + (s0 + l15) * HD;
  v16bf aq0 = load2x16(qrow + kh * 8,      qrow + 16 + kh * 8);
  v16bf aq1 = load2x16(qrow + 32 + kh * 8, qrow + 48 + kh * 8);

  v8f o[4] = {};
  float Mr[8], Lr[8];
#pragma unroll
  for (int r = 0; r < 8; ++r) { Mr[r] = -3.0e38f; Lr[r] = 0.f; }

  int kstart = s0 - LCTX; if (kstart < 0) kstart = 0;
  int kend   = s0 + 16 + LOOK; if (kend > S) kend = S;     // exclusive
  const int nsteps = (kend - kstart + 31) >> 5;
  const int npairs = (nsteps + 1) >> 1;   // pad to even #steps; tail step fully
                                          // masked -> alpha=1, p=0 (safe no-op)

  // K fragments: [f*2+half] covers keys kb+f*16.., hd-dim half*32..
  auto loadK = [&](int kb, v16bf bk[4]) {
#pragma unroll
    for (int f = 0; f < 2; ++f) {
      int key = kb + f * 16 + l15;
      int keyc = key < S ? key : S - 1;
      const bf16* kr = kp + keyc * HD;
      bk[f * 2 + 0] = load2x16(kr + kh * 8,      kr + 16 + kh * 8);       // k 0..31
      bk[f * 2 + 1] = load2x16(kr + 32 + kh * 8, kr + 48 + kh * 8);       // k 32..63
    }
  };

  // one 32-key online-softmax step using pre-loaded K fragments
  auto step = [&](int kb, const v16bf bk[4]) {
    v8f sc[2] = {};
#pragma unroll
    for (int f = 0; f < 2; ++f) {
      sc[f] = WMMA_BF16(aq0, bk[f * 2 + 0], sc[f]);
      sc[f] = WMMA_BF16(aq1, bk[f * 2 + 1], sc[f]);
    }
    // V fragments for this step: issue now so they fly during softmax VALU work
    v16bf bv[4];
#pragma unroll
    for (int t = 0; t < 4; ++t) {
      const bf16* vr = vh + (t * 16 + l15) * S + kb + kh * 8;
      bv[t] = load2x16(vr, vr + 16);
    }
    int key0 = kb + l15, key1 = kb + 16 + l15;
    float alpha[8];
#pragma unroll
    for (int r = 0; r < 8; ++r) {
      int qi = s0 + kh * 8 + r;
      float v0 = sc[0][r] * 0.125f;   // 1/sqrt(64)
      float v1 = sc[1][r] * 0.125f;
      bool a0 = (key0 <= qi + LOOK) && (key0 >= qi - LCTX) && (key0 < S);
      bool a1 = (key1 <= qi + LOOK) && (key1 >= qi - LCTX) && (key1 < S);
      v0 = a0 ? v0 : -1.0e30f;
      v1 = a1 ? v1 : -1.0e30f;
      float rm = fmaxf(v0, v1);
      rm = fmaxf(rm, __shfl_xor(rm, 1));
      rm = fmaxf(rm, __shfl_xor(rm, 2));
      rm = fmaxf(rm, __shfl_xor(rm, 4));
      rm = fmaxf(rm, __shfl_xor(rm, 8));
      float nm = fmaxf(Mr[r], rm);
      float al = __expf(Mr[r] - nm);
      Mr[r] = nm;
      float p0 = __expf(v0 - nm), p1 = __expf(v1 - nm);
      float rs = p0 + p1;
      rs += __shfl_xor(rs, 1);
      rs += __shfl_xor(rs, 2);
      rs += __shfl_xor(rs, 4);
      rs += __shfl_xor(rs, 8);
      Lr[r] = Lr[r] * al + rs;
      sc[0][r] = p0; sc[1][r] = p1;
      alpha[r] = al;
    }
#pragma unroll
    for (int t = 0; t < 4; ++t)
#pragma unroll
      for (int r = 0; r < 8; ++r) o[t][r] = o[t][r] * alpha[r];

    // relayout P: C-layout f32 -> A-layout bf16 through per-wave LDS tile
#pragma unroll
    for (int f = 0; f < 2; ++f)
#pragma unroll
      for (int r = 0; r < 8; ++r)
        lds[wl][kh * 8 + r][f * 16 + l15] = (bf16)sc[f][r];
    asm volatile("s_wait_dscnt 0" ::: "memory");
    const bf16* pr = &lds[wl][l15][0];
    v16bf ap = load2x16(pr + kh * 8, pr + 16 + kh * 8);

    // O += P(16x32) @ V(32x64); V^T layout makes B-frags contiguous
#pragma unroll
    for (int t = 0; t < 4; ++t) o[t] = WMMA_BF16(ap, bv[t], o[t]);
  };

  v16bf bkA[4], bkB[4];
  loadK(kstart, bkA);
  int kb = kstart;
  for (int p = 0; p < npairs; ++p) {
    loadK(kb + 32, bkB);   // prefetch: overlaps step A's softmax/WMMA
    step(kb, bkA);
    loadK(kb + 64, bkA);   // prefetch next pair (clamped addresses always valid)
    step(kb + 32, bkB);
    kb += 64;
  }

  // ---- normalize and write ctx[b,s,h*64+e] as bf16 ----
#pragma unroll
  for (int r = 0; r < 8; ++r) Lr[r] = 1.0f / Lr[r];
#pragma unroll
  for (int t = 0; t < 4; ++t)
#pragma unroll
    for (int r = 0; r < 8; ++r) {
      int s = s0 + kh * 8 + r;
      ctx[(bb * S + s) * D + h * HD + t * 16 + l15] = (bf16)(o[t][r] * Lr[r]);
    }
}

// ---------------- out projection: ctx @ W_out^T + b_out -> f32 ----------------
// 32x64 tile per wave; ping-pong pipelined like gemm_qkv.
__global__ __launch_bounds__(256) void gemm_out(
    const bf16* __restrict__ cx, const bf16* __restrict__ wo,
    const float* __restrict__ bo, float* __restrict__ out) {
  const int lane = threadIdx.x & 31;
  const int wid  = (blockIdx.x * blockDim.x + threadIdx.x) >> 5;
  const int NT = D / 64;                  // 16 column tiles
  const int mt = wid / NT, nt = wid % NT;
  const int l15 = lane & 15, kh = lane >> 4;

  const bf16* arow0 = cx + (mt * 32 + l15) * D;
  const bf16* arow1 = arow0 + 16 * D;
  const bf16* br[4];
#pragma unroll
  for (int j = 0; j < 4; ++j) br[j] = wo + (nt * 64 + j * 16 + l15) * D;

  v8f acc[2][4] = {};
  v16bf a0A, a1A, bA[4], a0B, a1B, bB[4];
  a0A = load2x16(arow0 + kh * 8, arow0 + 16 + kh * 8);
  a1A = load2x16(arow1 + kh * 8, arow1 + 16 + kh * 8);
#pragma unroll
  for (int j = 0; j < 4; ++j) bA[j] = load2x16(br[j] + kh * 8, br[j] + 16 + kh * 8);

  for (int k0 = 0; k0 < D; k0 += 64) {
    const int k1 = k0 + 32;
    a0B = load2x16(arow0 + k1 + kh * 8, arow0 + k1 + 16 + kh * 8);
    a1B = load2x16(arow1 + k1 + kh * 8, arow1 + k1 + 16 + kh * 8);
#pragma unroll
    for (int j = 0; j < 4; ++j)
      bB[j] = load2x16(br[j] + k1 + kh * 8, br[j] + k1 + 16 + kh * 8);
#pragma unroll
    for (int j = 0; j < 4; ++j) {
      acc[0][j] = WMMA_BF16(a0A, bA[j], acc[0][j]);
      acc[1][j] = WMMA_BF16(a1A, bA[j], acc[1][j]);
    }
    const int k2 = (k0 + 64 < D) ? (k0 + 64) : 0;
    a0A = load2x16(arow0 + k2 + kh * 8, arow0 + k2 + 16 + kh * 8);
    a1A = load2x16(arow1 + k2 + kh * 8, arow1 + k2 + 16 + kh * 8);
#pragma unroll
    for (int j = 0; j < 4; ++j)
      bA[j] = load2x16(br[j] + k2 + kh * 8, br[j] + k2 + 16 + kh * 8);
#pragma unroll
    for (int j = 0; j < 4; ++j) {
      acc[0][j] = WMMA_BF16(a0B, bB[j], acc[0][j]);
      acc[1][j] = WMMA_BF16(a1B, bB[j], acc[1][j]);
    }
  }
#pragma unroll
  for (int j = 0; j < 4; ++j) {
    int n = nt * 64 + j * 16 + l15;
    float bias = bo[n];
#pragma unroll
    for (int i = 0; i < 2; ++i)
#pragma unroll
      for (int r = 0; r < 8; ++r) {
        int m = mt * 32 + i * 16 + kh * 8 + r;
        out[m * D + n] = acc[i][j][r] + bias;
      }
  }
}

extern "C" void kernel_launch(void* const* d_in, const int* in_sizes, int n_in,
                              void* d_out, int out_size, void* d_ws, size_t ws_size,
                              hipStream_t stream) {
  (void)in_sizes; (void)n_in; (void)out_size; (void)ws_size;
  const float* x    = (const float*)d_in[0];
  const float* Win  = (const float*)d_in[1];
  const float* bin  = (const float*)d_in[2];
  const float* Wout = (const float*)d_in[3];
  const float* bout = (const float*)d_in[4];
  float* out = (float*)d_out;

  char* ws = (char*)d_ws;
  size_t off = 0;
  auto carve = [&](size_t bytes) -> void* {
    void* p = ws + off;
    off += (bytes + 255) & ~size_t(255);
    return p;
  };
  bf16* xb  = (bf16*)carve((size_t)M * D * 2);
  bf16* wib = (bf16*)carve((size_t)N3 * D * 2);
  bf16* wob = (bf16*)carve((size_t)D * D * 2);
  bf16* qbf = (bf16*)carve((size_t)M * D * 2);          // Q [B,H,S,HD]
  bf16* kbf = (bf16*)carve((size_t)M * D * 2);          // K [B,H,S,HD]
  bf16* vtb = (bf16*)carve((size_t)M * D * 2 + 4096);   // V^T [B,H,HD,S] + tail pad
  bf16* cxb = (bf16*)carve((size_t)M * D * 2);          // ctx [B,S,D]

  auto cvt = [&](const float* src, bf16* dst, int n) {
    int n4 = n / 4;
    cvt_bf16<<<(n4 + 255) / 256, 256, 0, stream>>>(src, dst, n4);
  };
  cvt(x, xb, M * D);
  cvt(Win, wib, N3 * D);
  cvt(Wout, wob, D * D);

  // 256 row-tiles(32) * 48 col-tiles(64) waves, 8 waves/block
  gemm_qkv<<<(M / 32) * (N3 / 64) / 8, 256, 0, stream>>>(xb, wib, bin, qbf, kbf, vtb);
  // B*H*(S/16) = 8192 query tiles, 8 waves/block
  attn<<<(Bx * H * (S / 16)) / 8, 256, 0, stream>>>(qbf, kbf, vtb, cxb);
  // 256 row-tiles(32) * 16 col-tiles(64) waves, 8 waves/block
  gemm_out<<<(M / 32) * (D / 64) / 8, 256, 0, stream>>>(cxb, wob, bout, out);
}
